// Attention_36283883716815
// MI455X (gfx1250) — matrix operand order repaired and checked by reading, not compiled, not run on hardware
//
#include <hip/hip_runtime.h>
#include <math.h>

// ---------------------------------------------------------------------------
// Problem constants (match the reference)
// ---------------------------------------------------------------------------
constexpr int Bc  = 4;
constexpr int Sc  = 1024;
constexpr int Cc  = 1024;
constexpr int HDc = 64;
constexpr int Hc  = Cc / HDc;       // 16
constexpr int Sfc = Sc + 1;         // 1025 (with sink token)
constexpr int ROTc = 32;
constexpr float EPSc = 1e-4f;

// ---------------------------------------------------------------------------
// Workspace layout (all chunks are 256-byte aligned by construction)
// ---------------------------------------------------------------------------
constexpr size_t OFF_SCAL = 0;                                   // 2 floats
constexpr size_t OFF_WQKV = 256;
constexpr size_t SZ_WQKV  = (size_t)3 * Cc * Cc * 2;             // bf16
constexpr size_t OFF_WOUT = OFF_WQKV + SZ_WQKV;
constexpr size_t SZ_WOUT  = (size_t)Cc * Cc * 2;
constexpr size_t OFF_XBF  = OFF_WOUT + SZ_WOUT;
constexpr size_t SZ_XBF   = (size_t)Bc * Sfc * Cc * 2;
constexpr size_t OFF_QKV  = OFF_XBF + SZ_XBF;
constexpr size_t SZ_QKV   = (size_t)Bc * Sfc * 3 * Cc * 4;       // fp32
constexpr size_t OFF_QBF  = OFF_QKV + SZ_QKV;
constexpr size_t SZ_HEAD  = (size_t)Bc * Hc * Sfc * HDc * 2;     // bf16
constexpr size_t OFF_KBF  = OFF_QBF + SZ_HEAD;
constexpr size_t OFF_VBF  = OFF_KBF + SZ_HEAD;
constexpr size_t OFF_H    = OFF_VBF + SZ_HEAD;
constexpr size_t SZ_H     = (size_t)Bc * Sc * Cc * 4;            // fp32
constexpr size_t OFF_HBF  = OFF_H + SZ_H;
constexpr size_t SZ_HBF   = (size_t)Bc * Sc * Cc * 2;
constexpr size_t WS_NEED  = OFF_HBF + SZ_HBF;                    // ~112 MB

// ---------------------------------------------------------------------------
// Vector types for WMMA fragments
// ---------------------------------------------------------------------------
typedef __attribute__((ext_vector_type(16))) __bf16 v16bf;
typedef __attribute__((ext_vector_type(8)))  __bf16 v8bf;
typedef __attribute__((ext_vector_type(8)))  float  v8f;

__device__ __forceinline__ v8f wmma_bf16(v16bf a, v16bf b, v8f c) {
    // D(16x16,f32) = A(16x32,bf16) x B(32x16,bf16) + C
    return __builtin_amdgcn_wmma_f32_16x16x32_bf16(
        /*neg_a=*/false, a, /*neg_b=*/false, b,
        /*c_mod=*/(short)0, c, /*reuse_a=*/false, /*reuse_b=*/false);
}

// A-matrix fragment (16x32 bf16) from row-major [rows x ld] at (row, k0).
// Contiguous k order used here (B matches): lane 0-15 holds row M=lane; element i -> K = half*16 + i.
__device__ __forceinline__ v16bf load_frag_a(const __bf16* P, int ld, int row,
                                             int k0, int lane) {
    int half = (lane >> 4) & 1;
    const __bf16* p = P + (size_t)row * ld + k0 + half * 16;
    v8bf lo = *(const v8bf*)(p);        // K = k0 + half*16 + [0..7]
    v8bf hi = *(const v8bf*)(p + 8);    // K = k0 + half*16 + 8 + [0..7]
    v16bf f;
#pragma unroll
    for (int i = 0; i < 8; ++i) { f[i] = lo[i]; f[i + 8] = hi[i]; }
    return f;
}

// B-matrix fragment (32x16 bf16, K x N) sourced from a row-major N x K matrix
// (i.e. this computes A @ source^T). Contiguous k order (matches load_frag_a): lane = column N, per-lane
// K = half*16 + i -> 16 contiguous bf16 = one 32-byte vector load.
__device__ __forceinline__ v16bf load_frag_bt(const __bf16* P, int ld, int nrow,
                                              int k0, int lane) {
    int half = (lane >> 4) & 1;
    return *(const v16bf*)(P + (size_t)nrow * ld + k0 + half * 16);
}

// Low 32 bits of a generic pointer into LDS == the LDS byte offset
// (addrspacecast AS3->flat places the AS3 offset in the low DWORD).
__device__ __forceinline__ uint32_t lds_off_u32(const void* p) {
    return (uint32_t)(uintptr_t)p;
}

// Fire-and-forget async copies (tracked by ASYNCcnt, no VGPR round trip).
__device__ __forceinline__ void async_g2l_b64(uint32_t lds, uint32_t voff,
                                              const void* base) {
    asm volatile("global_load_async_to_lds_b64 %0, %1, %2"
                 :: "v"(lds), "v"(voff), "s"((unsigned long long)(uintptr_t)base)
                 : "memory");
}
__device__ __forceinline__ void async_g2l_b128(uint32_t lds, uint32_t voff,
                                               const void* base) {
    asm volatile("global_load_async_to_lds_b128 %0, %1, %2"
                 :: "v"(lds), "v"(voff), "s"((unsigned long long)(uintptr_t)base)
                 : "memory");
}
__device__ __forceinline__ void wait_async0() {
    asm volatile("s_wait_asynccnt 0" ::: "memory");
}

// ---------------------------------------------------------------------------
// 0) zero accumulator scalars
// ---------------------------------------------------------------------------
__global__ void init_kernel(float* scalars) {
    scalars[0] = 0.0f;   // desired-norm accumulator
    scalars[1] = 0.0f;   // current-norm accumulator
}

// ---------------------------------------------------------------------------
// 1) weight row-normalization -> bf16.  W_eff = w_row / (||w_row|| + 32*EPS)
// ---------------------------------------------------------------------------
__global__ void wnorm_kernel(const float* __restrict__ W, __bf16* __restrict__ Wbf) {
    int row = blockIdx.x;
    int tid = threadIdx.x;
    const float* w = W + (size_t)row * Cc;
    float ss = 0.0f;
    for (int i = tid; i < Cc; i += 256) ss += w[i] * w[i];
    __shared__ float red[256];
    red[tid] = ss;
    __syncthreads();
    for (int s = 128; s > 0; s >>= 1) {
        if (tid < s) red[tid] += red[tid + s];
        __syncthreads();
    }
    float sc = 1.0f / (sqrtf(red[0]) + 32.0f * EPSc);
    for (int i = tid; i < Cc; i += 256)
        Wbf[(size_t)row * Cc + i] = (__bf16)(w[i] * sc);
}

// ---------------------------------------------------------------------------
// 2) build X_ext (append sink row) as bf16 + accumulate `desired` norm mean
// ---------------------------------------------------------------------------
__global__ void buildx_kernel(const float* __restrict__ x, const float* __restrict__ sink,
                              __bf16* __restrict__ xbf, float* __restrict__ scalars) {
    int row = blockIdx.x;                 // 0 .. B*Sf-1
    int b = row / Sfc, t = row % Sfc;
    int tid = threadIdx.x;
    const float* src = (t < Sc) ? (x + ((size_t)b * Sc + t) * Cc) : sink;
    float ss = 0.0f;
    for (int i = tid; i < Cc; i += 256) {
        float v = src[i];
        ss += v * v;
        xbf[(size_t)row * Cc + i] = (__bf16)v;
    }
    __shared__ float red[256];
    red[tid] = ss;
    __syncthreads();
    for (int s = 128; s > 0; s >>= 1) {
        if (tid < s) red[tid] += red[tid + s];
        __syncthreads();
    }
    if (tid == 0)
        atomicAdd(&scalars[0], sqrtf(red[0]) * (1.0f / (float)(Bc * Sfc)));
}

// ---------------------------------------------------------------------------
// 3) WMMA GEMM with double-buffered async-to-LDS staging:
//      D(MxN,f32) = A(MxK,bf16) @ Bt(NxK,bf16)^T
//    256 threads = 8 waves (2 M-subtiles x 4 N-subtiles) per 32x64 block tile.
//    Per K-step (32): every thread issues one async b64 (A tile, 2 KB) and one
//    async b128 (B tile, 4 KB) into the *next* buffer, does one WMMA from the
//    *current* LDS tiles, drains ASYNCcnt, barriers. Stage s+1 writes the
//    buffer last read at stage s-1, whose readers all passed the previous
//    barrier -> no WAR race. Edge M rows: clamp-on-load, guard-on-store so
//    control flow stays wave-uniform (EXEC all ones for WMMA).
// ---------------------------------------------------------------------------
__device__ __forceinline__ void async_tile_copy(
    const __bf16* __restrict__ A, const __bf16* __restrict__ Bt,
    int M, int N, int K, int m0, int n0, int k0,
    uint32_t ldsA, uint32_t ldsB, int tid) {
    {   // A tile: 32 rows x 32 k (2 KB), 8 bytes/thread
        int row = tid >> 3, part = tid & 7;
        int ar = m0 + row; if (ar > M - 1) ar = M - 1;
        uint32_t voff = (uint32_t)(((size_t)ar * K + k0 + part * 4) * 2);
        async_g2l_b64(ldsA + (uint32_t)(row * 64 + part * 8), voff, A);
    }
    {   // B tile: 64 rows x 32 k (4 KB), 16 bytes/thread
        int row = tid >> 2, part = tid & 3;
        int br = n0 + row; if (br > N - 1) br = N - 1;
        uint32_t voff = (uint32_t)(((size_t)br * K + k0 + part * 8) * 2);
        async_g2l_b128(ldsB + (uint32_t)(row * 64 + part * 16), voff, Bt);
    }
}

__global__ void __launch_bounds__(256)
gemm_bf16_nt(const __bf16* __restrict__ A, const __bf16* __restrict__ Bt,
             float* __restrict__ D, int M, int N, int K) {
    __shared__ __bf16 ldsA[2][32 * 32];
    __shared__ __bf16 ldsB[2][64 * 32];
    int tid  = threadIdx.x;
    int lane = tid & 31;
    int wave = tid >> 5;              // 0..7
    int half = lane >> 4, col = lane & 15;
    int wm = wave >> 2;               // 0..1  M sub-tile
    int wn = wave & 3;                // 0..3  N sub-tile
    int m0 = blockIdx.x * 32;
    int n0 = blockIdx.y * 64;

    uint32_t lA[2] = { lds_off_u32(&ldsA[0][0]), lds_off_u32(&ldsA[1][0]) };
    uint32_t lB[2] = { lds_off_u32(&ldsB[0][0]), lds_off_u32(&ldsB[1][0]) };

    int nsteps = K / 32;
    async_tile_copy(A, Bt, M, N, K, m0, n0, 0, lA[0], lB[0], tid);
    wait_async0();
    __syncthreads();

    v8f c = {};
    for (int s = 0; s < nsteps; ++s) {
        int cur = s & 1;
        if (s + 1 < nsteps)
            async_tile_copy(A, Bt, M, N, K, m0, n0, (s + 1) * 32,
                            lA[cur ^ 1], lB[cur ^ 1], tid);

        const __bf16* tA = &ldsA[cur][0];
        const __bf16* tB = &ldsB[cur][0];
        v16bf a = load_frag_a(tA, 32, wm * 16 + col, 0, lane);
        v16bf b = load_frag_bt(tB, 32, wn * 16 + col, 0, lane);
        c = wmma_bf16(a, b, c);

        wait_async0();      // drain stage s+1 copies (compute overlapped them)
        __syncthreads();
    }

#pragma unroll
    for (int r = 0; r < 8; ++r) {
        int m = m0 + wm * 16 + r + 8 * half;
        if (m < M) D[(size_t)m * N + (n0 + wn * 16 + col)] = c[r];
    }
}

// ---------------------------------------------------------------------------
// 4) QKV post-processing: scale, RoPE (non-sink, first 32 dims), L2-normalize
//    q,k; emit head-major bf16 Q/K/V of shape [B][H][Sf][64].
// ---------------------------------------------------------------------------
__global__ void qkv_post_kernel(const float* __restrict__ qkv, const float* __restrict__ re,
                                __bf16* __restrict__ qbf, __bf16* __restrict__ kbf,
                                __bf16* __restrict__ vbf) {
    int idx = blockIdx.x;                  // b * Sf * H + t * H + h
    int h = idx % Hc;
    int t = (idx / Hc) % Sfc;
    int b = idx / (Hc * Sfc);
    int d = threadIdx.x;                   // 0..63
    const float scale = 0.3535533905932738f;  // 64^-0.25

    size_t rowbase = ((size_t)b * Sfc + t) * (3 * Cc);
    float qd = qkv[rowbase + h * HDc + d] * scale;
    float kd = qkv[rowbase + Cc + h * HDc + d] * scale;
    float vd = qkv[rowbase + 2 * Cc + h * HDc + d];

    if (t < Sc && d < ROTc) {
        int pd = (d < 16) ? d + 16 : d - 16;
        float sgn = (d < 16) ? -1.0f : 1.0f;
        float f = re[(size_t)t * ROTc + d];
        float cf = cosf(f), sf = sinf(f);
        float qp = qkv[rowbase + h * HDc + pd] * scale;
        float kp = qkv[rowbase + Cc + h * HDc + pd] * scale;
        qd = qd * cf + sgn * qp * sf;
        kd = kd * cf + sgn * kp * sf;
    }

    __shared__ float sq[64], sk[64];
    sq[d] = qd * qd;
    sk[d] = kd * kd;
    __syncthreads();
    for (int s = 32; s > 0; s >>= 1) {
        if (d < s) { sq[d] += sq[d + s]; sk[d] += sk[d + s]; }
        __syncthreads();
    }
    float qinv = rsqrtf(sq[0]);
    float kinv = rsqrtf(sk[0]);

    size_t obase = (((size_t)b * Hc + h) * Sfc + t) * HDc + d;
    qbf[obase] = (__bf16)(qd * qinv);
    kbf[obase] = (__bf16)(kd * kinv);
    vbf[obase] = (__bf16)vd;
}

// ---------------------------------------------------------------------------
// 5) flash attention with causal+sink mask (one wave per 16-row q-tile).
// ---------------------------------------------------------------------------
constexpr int QT = (Sfc + 15) / 16;         // 65
constexpr int TOTALW = Bc * Hc * QT;        // 4160 (divisible by 4 waves/block)

__global__ void attn_kernel(const __bf16* __restrict__ qbf, const __bf16* __restrict__ kbf,
                            const __bf16* __restrict__ vbf, float* __restrict__ Hout) {
    __shared__ __bf16 ldsP[4][16 * 32];
    int wave = threadIdx.x >> 5;
    int lane = threadIdx.x & 31;
    int half = lane >> 4, col = lane & 15;
    int wg = blockIdx.x * 4 + wave;

    int qt = wg % QT;
    int bh = wg / QT;
    int hh = bh % Hc;
    int b  = bh / Hc;

    const __bf16* Qp = qbf + (size_t)(b * Hc + hh) * Sfc * HDc;
    const __bf16* Kp = kbf + (size_t)(b * Hc + hh) * Sfc * HDc;
    const __bf16* Vp = vbf + (size_t)(b * Hc + hh) * Sfc * HDc;
    __bf16* myP = ldsP[wave];

    int qr = qt * 16 + col; if (qr > Sfc - 1) qr = Sfc - 1;
    v16bf qf0 = load_frag_a(Qp, HDc, qr, 0, lane);
    v16bf qf1 = load_frag_a(Qp, HDc, qr, 32, lane);

    float m_run[8], l_run[8];
    v8f acc[4];
#pragma unroll
    for (int r = 0; r < 8; ++r) { m_run[r] = -1e30f; l_run[r] = 0.0f; }
#pragma unroll
    for (int ch = 0; ch < 4; ++ch) { v8f z = {}; acc[ch] = z; }

    for (int j0 = 0; j0 < Sfc; j0 += 32) {
        int k0c = j0 + col;       if (k0c > Sfc - 1) k0c = Sfc - 1;
        int k1c = j0 + 16 + col;  if (k1c > Sfc - 1) k1c = Sfc - 1;
        v16bf kA0 = load_frag_bt(Kp, HDc, k0c, 0, lane);
        v16bf kB0 = load_frag_bt(Kp, HDc, k0c, 32, lane);
        v16bf kA1 = load_frag_bt(Kp, HDc, k1c, 0, lane);
        v16bf kB1 = load_frag_bt(Kp, HDc, k1c, 32, lane);

        v8f c0 = {}, c1 = {};
        c0 = wmma_bf16(qf0, kA0, c0);
        c0 = wmma_bf16(qf1, kB0, c0);
        c1 = wmma_bf16(qf0, kA1, c1);
        c1 = wmma_bf16(qf1, kB1, c1);

#pragma unroll
        for (int r = 0; r < 8; ++r) {
            int i = qt * 16 + r + 8 * half;
            int j0c = j0 + col, j1c = j0 + 16 + col;
            bool sinkrow = (i == Sfc - 1);
            bool ok0 = (i < Sfc) && (j0c < Sfc) && (sinkrow || j0c <= i || j0c == Sfc - 1);
            bool ok1 = (i < Sfc) && (j1c < Sfc) && (sinkrow || j1c <= i || j1c == Sfc - 1);
            float s0 = ok0 ? c0[r] : -1e30f;
            float s1 = ok1 ? c1[r] : -1e30f;

            float mx = fmaxf(s0, s1);
            mx = fmaxf(mx, __shfl_xor(mx, 1));
            mx = fmaxf(mx, __shfl_xor(mx, 2));
            mx = fmaxf(mx, __shfl_xor(mx, 4));
            mx = fmaxf(mx, __shfl_xor(mx, 8));

            float mnew = fmaxf(m_run[r], mx);
            float alpha = __expf(m_run[r] - mnew);
            m_run[r] = mnew;
            float p0 = __expf(s0 - mnew);
            float p1 = __expf(s1 - mnew);
            float ps = p0 + p1;
            ps += __shfl_xor(ps, 1);
            ps += __shfl_xor(ps, 2);
            ps += __shfl_xor(ps, 4);
            ps += __shfl_xor(ps, 8);
            l_run[r] = l_run[r] * alpha + ps;
            acc[0][r] *= alpha; acc[1][r] *= alpha;
            acc[2][r] *= alpha; acc[3][r] *= alpha;

            // P in C-layout -> LDS (per-wave region, intra-wave ordered by DScnt)
            myP[(r + 8 * half) * 32 + col]      = (__bf16)p0;
            myP[(r + 8 * half) * 32 + 16 + col] = (__bf16)p1;
        }

        // re-load P as an A-fragment (16 rows x 32 keys)
        v16bf pf = load_frag_a(myP, 32, col, 0, lane);

#pragma unroll
        for (int ch = 0; ch < 4; ++ch) {
            int dcol = ch * 16 + col;
            v16bf vf;
#pragma unroll
            for (int i2 = 0; i2 < 16; ++i2) {
                int j = j0 + half * 16 + i2;
                if (j > Sfc - 1) j = Sfc - 1;   // P is 0 there -> no contribution
                vf[i2] = Vp[(size_t)j * HDc + dcol];
            }
            acc[ch] = wmma_bf16(pf, vf, acc[ch]);
        }
    }

#pragma unroll
    for (int r = 0; r < 8; ++r) {
        int i = qt * 16 + r + 8 * half;
        if (i < Sc) {
            float inv = 1.0f / l_run[r];
            size_t base = ((size_t)b * Sc + i) * Cc + hh * HDc;
            Hout[base +  0 + col] = acc[0][r] * inv;
            Hout[base + 16 + col] = acc[1][r] * inv;
            Hout[base + 32 + col] = acc[2][r] * inv;
            Hout[base + 48 + col] = acc[3][r] * inv;
        }
    }
}

// ---------------------------------------------------------------------------
// 6) accumulate `current` = mean token norm of h
// ---------------------------------------------------------------------------
__global__ void hnorm_kernel(const float* __restrict__ h, float* __restrict__ scalars) {
    int row = blockIdx.x;                 // 0 .. B*S-1
    int tid = threadIdx.x;
    const float* p = h + (size_t)row * Cc;
    float ss = 0.0f;
    for (int i = tid; i < Cc; i += 256) { float v = p[i]; ss += v * v; }
    __shared__ float red[256];
    red[tid] = ss;
    __syncthreads();
    for (int s = 128; s > 0; s >>= 1) {
        if (tid < s) red[tid] += red[tid + s];
        __syncthreads();
    }
    if (tid == 0)
        atomicAdd(&scalars[1], sqrtf(red[0]) * (1.0f / (float)(Bc * Sc)));
}

// ---------------------------------------------------------------------------
// 7) h * (desired/current) -> bf16
// ---------------------------------------------------------------------------
__global__ void hscale_kernel(const float* __restrict__ h, const float* __restrict__ scalars,
                              __bf16* __restrict__ hbf) {
    float s = scalars[0] / scalars[1];
    size_t i = (size_t)blockIdx.x * blockDim.x + threadIdx.x;
    hbf[i] = (__bf16)(h[i] * s);
}

// ---------------------------------------------------------------------------
// Host-side launcher
// ---------------------------------------------------------------------------
extern "C" void kernel_launch(void* const* d_in, const int* in_sizes, int n_in,
                              void* d_out, int out_size, void* d_ws, size_t ws_size,
                              hipStream_t stream) {
    (void)in_sizes; (void)n_in; (void)out_size;
    if (ws_size < WS_NEED) return;   // deterministic bail-out if scratch too small

    const float* x      = (const float*)d_in[0];
    const float* re     = (const float*)d_in[1];
    // d_in[2] = attn_mask (computed analytically, unused)
    const float* w_qkv  = (const float*)d_in[3];
    const float* w_out  = (const float*)d_in[4];
    const float* sink   = (const float*)d_in[5];
    float* out = (float*)d_out;

    char* ws = (char*)d_ws;
    float*  scalars = (float*)(ws + OFF_SCAL);
    __bf16* wqkv_bf = (__bf16*)(ws + OFF_WQKV);
    __bf16* wout_bf = (__bf16*)(ws + OFF_WOUT);
    __bf16* xbf     = (__bf16*)(ws + OFF_XBF);
    float*  qkvf    = (float*)(ws + OFF_QKV);
    __bf16* qbf     = (__bf16*)(ws + OFF_QBF);
    __bf16* kbf     = (__bf16*)(ws + OFF_KBF);
    __bf16* vbf     = (__bf16*)(ws + OFF_VBF);
    float*  hbuf    = (float*)(ws + OFF_H);
    __bf16* hbf     = (__bf16*)(ws + OFF_HBF);

    init_kernel<<<1, 1, 0, stream>>>(scalars);
    wnorm_kernel<<<3 * Cc, 256, 0, stream>>>(w_qkv, wqkv_bf);
    wnorm_kernel<<<Cc, 256, 0, stream>>>(w_out, wout_bf);
    buildx_kernel<<<Bc * Sfc, 256, 0, stream>>>(x, sink, xbf, scalars);

    // QKV GEMM: (4100 x 1024) @ (3072 x 1024)^T -> (4100 x 3072) fp32
    {
        int M = Bc * Sfc, N = 3 * Cc, K = Cc;
        dim3 grid((M + 31) / 32, N / 64);
        gemm_bf16_nt<<<grid, 256, 0, stream>>>(xbf, wqkv_bf, qkvf, M, N, K);
    }

    qkv_post_kernel<<<Bc * Sfc * Hc, 64, 0, stream>>>(qkvf, re, qbf, kbf, vbf);

    attn_kernel<<<TOTALW / 4, 128, 0, stream>>>(qbf, kbf, vbf, hbuf);

    hnorm_kernel<<<Bc * Sc, 256, 0, stream>>>(hbuf, scalars);
    hscale_kernel<<<(Bc * Sc * Cc) / 256, 256, 0, stream>>>(hbuf, scalars, hbf);

    // Output GEMM: (4096 x 1024) @ (1024 x 1024)^T -> d_out fp32
    {
        int M = Bc * Sc, N = Cc, K = Cc;
        dim3 grid(M / 32, N / 64);
        gemm_bf16_nt<<<grid, 256, 0, stream>>>(hbf, wout_bf, out, M, N, K);
    }
}
